// CustomQuanDense_36283883717255
// MI455X (gfx1250) — 8-bit matrix operand order repaired and checked by reading, compiled, not run on hardware
//
#include <hip/hip_runtime.h>
#include <stdint.h>

typedef int v8i  __attribute__((ext_vector_type(8)));
typedef int vint4 __attribute__((vector_size(16)));
typedef __attribute__((address_space(1))) vint4 gvint4;
typedef __attribute__((address_space(3))) vint4 lvint4;

#define QMAX_F   127.0f
#define M_DIM    8192
#define K_DIM    4096
#define N_DIM    4096

#define BM       64
#define BN       128
#define BK       64
#define LDST     80

__device__ __forceinline__ void async_cp_b128(void* lds_dst, const void* gsrc)
{
    __builtin_amdgcn_global_load_async_to_lds_b128(
        (gvint4*)(vint4*)(void*)gsrc,
        (lvint4*)(vint4*)lds_dst,
        0, 0);
}

__device__ __forceinline__ void wait_async0()
{
    asm volatile("s_wait_asynccnt 0x0" ::: "memory");
}

__global__ __launch_bounds__(256) void absmax_f32(const float4* __restrict__ x,
                                                  int n4, unsigned* __restrict__ out)
{
    float m = 0.0f;
    int stride = gridDim.x * blockDim.x;
    for (int i = blockIdx.x * blockDim.x + threadIdx.x; i < n4; i += stride) {
        float4 v = x[i];
        m = fmaxf(m, fabsf(v.x));
        m = fmaxf(m, fabsf(v.y));
        m = fmaxf(m, fabsf(v.z));
        m = fmaxf(m, fabsf(v.w));
    }
#pragma unroll
    for (int o = 16; o > 0; o >>= 1)
        m = fmaxf(m, __shfl_xor(m, o, 32));
    if ((threadIdx.x & 31) == 0)
        atomicMax(out, __float_as_uint(m));
}

__global__ __launch_bounds__(256) void quant_i8(const float4* __restrict__ x,
                                                int* __restrict__ q, int n4,
                                                const unsigned* __restrict__ amax_bits)
{
    float a     = __uint_as_float(*amax_bits);
    float scale = fmaxf(a * (1.0f / QMAX_F), 1e-8f);
    float inv   = 1.0f / scale;
    int stride  = gridDim.x * blockDim.x;
    for (int i = blockIdx.x * blockDim.x + threadIdx.x; i < n4; i += stride) {
        float4 v = x[i];
        int b0 = (int)__builtin_rintf(fminf(fmaxf(v.x * inv, -QMAX_F), QMAX_F));
        int b1 = (int)__builtin_rintf(fminf(fmaxf(v.y * inv, -QMAX_F), QMAX_F));
        int b2 = (int)__builtin_rintf(fminf(fmaxf(v.z * inv, -QMAX_F), QMAX_F));
        int b3 = (int)__builtin_rintf(fminf(fmaxf(v.w * inv, -QMAX_F), QMAX_F));
        q[i] = (b0 & 255) | ((b1 & 255) << 8) | ((b2 & 255) << 16) | ((b3 & 255) << 24);
    }
}

__global__ __launch_bounds__(256) void gemm_i8_wmma(const char* __restrict__ qx,
                                                    const char* __restrict__ qw,
                                                    float* __restrict__ out,
                                                    const unsigned* __restrict__ scales)
{
    __shared__ char lA[2][BM * LDST];
    __shared__ char lB[2][BN * LDST];

    const int tid   = threadIdx.x;
    const int lane  = tid & 31;
    const int wave  = tid >> 5;
    const int waveM = wave & 1;
    const int waveN = wave >> 1;
    const int sel   = lane >> 4;
    const int l16   = lane & 15;

    const int blockN0 = blockIdx.x * BN;
    const int blockM0 = blockIdx.y * BM;

    const float ax = __uint_as_float(scales[0]);
    const float aw = __uint_as_float(scales[1]);
    const float dq = fmaxf(ax * (1.0f / QMAX_F), 1e-8f) *
                     fmaxf(aw * (1.0f / QMAX_F), 1e-8f);

    v8i acc[2][2];
#pragma unroll
    for (int tm = 0; tm < 2; ++tm)
#pragma unroll
        for (int tn = 0; tn < 2; ++tn)
            acc[tm][tn] = (v8i){0, 0, 0, 0, 0, 0, 0, 0};

    const int arow = tid >> 2;
    const int acol = (tid & 3) << 4;

    auto stage = [&](int buf, int k0) {
        async_cp_b128(&lA[buf][arow * LDST + acol],
                      &qx[(size_t)(blockM0 + arow) * K_DIM + k0 + acol]);
#pragma unroll
        for (int i = 0; i < 2; ++i) {
            int idx  = tid + i * 256;
            int brow = idx >> 2;
            int bcol = (idx & 3) << 4;
            async_cp_b128(&lB[buf][brow * LDST + bcol],
                          &qw[(size_t)(blockN0 + brow) * K_DIM + k0 + bcol]);
        }
    };

    union Frag { v8i v; unsigned long long d[4]; };

    stage(0, 0);
    wait_async0();
    __syncthreads();

    const int NITER = K_DIM / BK;
    for (int it = 0; it < NITER; ++it) {
        const int b = it & 1;

        Frag afr[2], bfr[2];
#pragma unroll
        for (int t = 0; t < 2; ++t) {
            const char* ap = &lA[b][(waveM * 32 + t * 16 + l16) * LDST + sel * 8];
            afr[t].d[0] = *(const unsigned long long*)(ap + 0);
            afr[t].d[1] = *(const unsigned long long*)(ap + 16);
            afr[t].d[2] = *(const unsigned long long*)(ap + 32);
            afr[t].d[3] = *(const unsigned long long*)(ap + 48);
            const char* bp = &lB[b][(waveN * 32 + t * 16 + l16) * LDST + sel * 8];
            bfr[t].d[0] = *(const unsigned long long*)(bp + 0);
            bfr[t].d[1] = *(const unsigned long long*)(bp + 16);
            bfr[t].d[2] = *(const unsigned long long*)(bp + 32);
            bfr[t].d[3] = *(const unsigned long long*)(bp + 48);
        }

        if (it + 1 < NITER)
            stage(b ^ 1, (it + 1) * BK);

#pragma unroll
        for (int tm = 0; tm < 2; ++tm)
#pragma unroll
            for (int tn = 0; tn < 2; ++tn)
                acc[tm][tn] = __builtin_amdgcn_wmma_i32_16x16x64_iu8(
                    true, afr[tm].v, true, bfr[tn].v, acc[tm][tn], false, false);

        wait_async0();
        __syncthreads();
    }

#pragma unroll
    for (int tm = 0; tm < 2; ++tm) {
#pragma unroll
        for (int tn = 0; tn < 2; ++tn) {
            int col = blockN0 + waveN * 32 + tn * 16 + l16;
#pragma unroll
            for (int j = 0; j < 8; ++j) {
                int row = blockM0 + waveM * 32 + tm * 16 + sel * 8 + j;
                out[(size_t)row * N_DIM + col] = dq * (float)acc[tm][tn][j];
            }
        }
    }
}

extern "C" void kernel_launch(void* const* d_in, const int* in_sizes, int n_in,
                              void* d_out, int out_size, void* d_ws, size_t ws_size,
                              hipStream_t stream)
{
    const float* x = (const float*)d_in[0];
    const float* w = (const float*)d_in[1];
    float* out = (float*)d_out;

    unsigned* scales = (unsigned*)d_ws;
    char* qx = (char*)d_ws + 256;
    char* qw = qx + (size_t)M_DIM * K_DIM;

    (void)hipMemsetAsync(scales, 0, 256, stream);

    const int n4x = (M_DIM * K_DIM) / 4;
    const int n4w = (N_DIM * K_DIM) / 4;

    absmax_f32<<<2048, 256, 0, stream>>>((const float4*)x, n4x, &scales[0]);
    absmax_f32<<<2048, 256, 0, stream>>>((const float4*)w, n4w, &scales[1]);
    quant_i8<<<2048, 256, 0, stream>>>((const float4*)x, (int*)qx, n4x, &scales[0]);
    quant_i8<<<2048, 256, 0, stream>>>((const float4*)w, (int*)qw, n4w, &scales[1]);

    dim3 grid(N_DIM / BN, M_DIM / BM);
    gemm_i8_wmma<<<grid, 256, 0, stream>>>(qx, qw, out, scales);
}
